// MLA_91173565759557
// MI455X (gfx1250) — matrix operand order repaired and checked by reading, not compiled, not run on hardware
//
#include <hip/hip_runtime.h>

// ---------------------------------------------------------------------------
// MLA forward for gfx1250 (MI455X): bf16 WMMA GEMMs + flash attention.
// ---------------------------------------------------------------------------

typedef __attribute__((ext_vector_type(16))) __bf16        bf16x16;
typedef __attribute__((ext_vector_type(8)))  float         f32x8;
typedef __attribute__((ext_vector_type(4)))  unsigned int  u32x4;
typedef unsigned short u16;

#define T_SEQ 2048
#define DM    2048
#define DCQ   1536
#define DC    512
#define NH    32
#define DH    64
#define DQK   128   // DH + D_ROPE

__device__ __forceinline__ u16 f32_to_bf16(float f) {
  unsigned int u = __builtin_bit_cast(unsigned int, f);
  u += 0x7FFFu + ((u >> 16) & 1u);   // round to nearest even
  return (u16)(u >> 16);
}

union FragCast { u32x4 q[2]; bf16x16 v; };

// A-operand 16x32 bf16 fragment (ISA 7.12.2): lane<16 row=lane elems K{0..7,16..23},
// lane>=16 same row elems K{8..15,24..31}. Source row-major, K contiguous.
__device__ __forceinline__ bf16x16 load_frag_a(const u16* p, int ld, int lane) {
  const int r  = lane & 15;
  const int kh = (lane >> 4) & 1;
  const u16* q = p + (size_t)r * ld + kh * 8;
  FragCast u;
  u.q[0] = *reinterpret_cast<const u32x4*>(q);
  u.q[1] = *reinterpret_cast<const u32x4*>(q + 16);
  return u.v;
}

// B-operand 32x16 bf16 fragment: lane col n=lane&15, lane<16 elems K{0..7,16..23},
// lane>=16 elems K{8..15,24..31}. Source "W" stored (N x K) row-major => C = A*W^T.
__device__ __forceinline__ bf16x16 load_frag_b(const u16* p, int ld, int lane) {
  const int n = lane & 15;
  const int g = (lane >> 4) & 1;
  const u16* q = p + (size_t)n * ld + g * 8;
  FragCast u;
  u.q[0] = *reinterpret_cast<const u32x4*>(q);
  u.q[1] = *reinterpret_cast<const u32x4*>(q + 16);
  return u.v;
}

__device__ __forceinline__ f32x8 wmma_bf16(bf16x16 a, bf16x16 b, f32x8 c) {
  return __builtin_amdgcn_wmma_f32_16x16x32_bf16(false, a, false, b,
                                                 (short)0, c, false, false);
}

// ---------------------------------------------------------------------------
// fp32 -> bf16 elementwise conversion
// ---------------------------------------------------------------------------
__global__ void cvt_bf16_kernel(const float* __restrict__ in,
                                u16* __restrict__ out, int n) {
  int i = blockIdx.x * blockDim.x + threadIdx.x;
  if (i < n) out[i] = f32_to_bf16(in[i]);
}

// ---------------------------------------------------------------------------
// C(MxN) = A(MxK) * W(NxK)^T, bf16 in, fp32 accum.
// Block: 256 threads = 8 waves arranged 4 (rows) x 2 (cols).
// Wave tile: 32 x (16*NW); block tile: 128 x (32*NW).  NW=4 for large N
// (higher WMMA-per-load density), NW=2 for the N=64 rope projections.
// mode 0: fp32 row-major to Cf
// mode 1: bf16 to Cb in Q'/K' layout  [h][t][d] (d<64), h=n>>6, d=n&63, t=m
// mode 2: bf16 to Cb in V^T layout    [h][d][t]
// ---------------------------------------------------------------------------
template <int NW>
__global__ __launch_bounds__(256)
void gemm_bf16_kernel(const u16* __restrict__ A, const u16* __restrict__ W,
                      float* __restrict__ Cf, u16* __restrict__ Cb,
                      int M, int N, int K, int mode) {
  const int lane  = threadIdx.x & 31;
  const int wave  = threadIdx.x >> 5;
  const int row0  = blockIdx.y * 128 + (wave >> 1) * 32;
  const int col0  = blockIdx.x * (32 * NW) + (wave & 1) * (16 * NW);
  f32x8 acc[2][NW];
#pragma unroll
  for (int im = 0; im < 2; ++im)
#pragma unroll
    for (int jn = 0; jn < NW; ++jn) acc[im][jn] = (f32x8){};

  const int ar = lane & 15;           // row within a fragment this lane touches
  for (int k0 = 0; k0 < K; k0 += 32) {
    // branchless prefetch 2 K-steps (128B) ahead -> global_prefetch_b8
    int kpre = (k0 + 64 < K) ? (k0 + 64) : k0;   // uniform select, no branch
    __builtin_prefetch(A + (size_t)(row0 + ar) * K + kpre, 0, 1);
    __builtin_prefetch(W + (size_t)(col0 + ar) * K + kpre, 0, 1);
    bf16x16 a0 = load_frag_a(A + (size_t)row0 * K + k0, K, lane);
    bf16x16 a1 = load_frag_a(A + (size_t)(row0 + 16) * K + k0, K, lane);
    bf16x16 b[NW];
#pragma unroll
    for (int jn = 0; jn < NW; ++jn)
      b[jn] = load_frag_b(W + (size_t)(col0 + 16 * jn) * K + k0, K, lane);
#pragma unroll
    for (int jn = 0; jn < NW; ++jn) {
      acc[0][jn] = wmma_bf16(a0, b[jn], acc[0][jn]);
      acc[1][jn] = wmma_bf16(a1, b[jn], acc[1][jn]);
    }
  }
  const int g = lane >> 4, c = lane & 15;
#pragma unroll
  for (int im = 0; im < 2; ++im)
#pragma unroll
    for (int jn = 0; jn < NW; ++jn)
#pragma unroll
      for (int i = 0; i < 8; ++i) {
        int m = row0 + im * 16 + i + 8 * g;
        int n = col0 + jn * 16 + c;
        float v = acc[im][jn][i];
        if (mode == 0) {
          Cf[(size_t)m * N + n] = v;
        } else if (mode == 1) {
          Cb[(((size_t)(n >> 6)) * T_SEQ + m) * DQK + (n & 63)] = f32_to_bf16(v);
        } else {
          Cb[(((size_t)(n >> 6)) * DH + (n & 63)) * T_SEQ + m] = f32_to_bf16(v);
        }
      }
}

// ---------------------------------------------------------------------------
// RMSNorm per row: y = x * rsqrt(mean(x^2)+eps) * w ; bf16 (+optional fp32) out
// ---------------------------------------------------------------------------
__global__ __launch_bounds__(256)
void rmsnorm_kernel(const float* __restrict__ in, const float* __restrict__ w,
                    float* __restrict__ out_f32, u16* __restrict__ out_b16,
                    int D) {
  const int row = blockIdx.x;
  const float* x = in + (size_t)row * D;
  float ss = 0.f;
  for (int d = threadIdx.x; d < D; d += blockDim.x) { float v = x[d]; ss += v * v; }
  __shared__ float red[256];
  red[threadIdx.x] = ss;
  __syncthreads();
  for (int s = 128; s > 0; s >>= 1) {
    if ((int)threadIdx.x < s) red[threadIdx.x] += red[threadIdx.x + s];
    __syncthreads();
  }
  const float scale = rsqrtf(red[0] / (float)D + 1.1920928955078125e-7f);
  for (int d = threadIdx.x; d < D; d += blockDim.x) {
    float v = x[d] * scale * w[d];
    if (out_f32) out_f32[(size_t)row * D + d] = v;
    out_b16[(size_t)row * D + d] = f32_to_bf16(v);
  }
}

// ---------------------------------------------------------------------------
// RoPE on (T x 64) fp32; broadcast bf16 result into rope half of Q'/K' for all
// heads. For K also emit k_rope_cache (64 x 32 fp32) at t==0.
// ---------------------------------------------------------------------------
__global__ void rope_kernel(const float* __restrict__ in, u16* __restrict__ dst,
                            float* __restrict__ cache) {
  const int t = blockIdx.x;
  const int i = threadIdx.x;  // pair index 0..31
  const float inv = __powf(10000.0f, -(2.0f * (float)i) / 64.0f);
  float s, c;
  __sincosf((float)t * inv, &s, &c);
  const float x0 = in[t * 64 + 2 * i];
  const float x1 = in[t * 64 + 2 * i + 1];
  const float o0 = x0 * c - x1 * s;
  const float o1 = x0 * s + x1 * c;
  const u16 b0 = f32_to_bf16(o0), b1 = f32_to_bf16(o1);
  for (int h = 0; h < NH; ++h) {
    size_t base = ((size_t)h * T_SEQ + t) * DQK + DH;
    dst[base + 2 * i]     = b0;
    dst[base + 2 * i + 1] = b1;
  }
  if (cache != nullptr && t == 0) {
    for (int h = 0; h < NH; ++h) {
      cache[(2 * i) * NH + h]     = o0;
      cache[(2 * i + 1) * NH + h] = o1;
    }
  }
}

// ---------------------------------------------------------------------------
// Flash attention, d=128 (content 64 + rope 64 folded in), dv=64, no mask.
// Block = 8 waves; each wave owns 16 query rows; key tiles of 32.
// ---------------------------------------------------------------------------
__global__ __launch_bounds__(256)
void flash_attn_kernel(const u16* __restrict__ Qb, const u16* __restrict__ Kb,
                       const u16* __restrict__ Vt, u16* __restrict__ Ob) {
  __shared__ __align__(16) u16 pshm[8][16 * 32];
  const int lane = threadIdx.x & 31;
  const int wave = threadIdx.x >> 5;
  const int h  = blockIdx.y;
  const int q0 = blockIdx.x * 128 + wave * 16;
  const int g = lane >> 4, cc = lane & 15;
  const u16* Qp = Qb + ((size_t)h * T_SEQ + q0) * DQK;
  bf16x16 qf[4];
#pragma unroll
  for (int kk = 0; kk < 4; ++kk) qf[kk] = load_frag_a(Qp + kk * 32, DQK, lane);

  f32x8 o0 = {}, o1 = {}, o2 = {}, o3 = {};
  float m_i[8], l_i[8];
#pragma unroll
  for (int i = 0; i < 8; ++i) { m_i[i] = -1e30f; l_i[i] = 0.f; }
  u16* ps = pshm[wave];
  const float scale = 0.08838834764831845f;  // 1/sqrt(128)

  for (int j0 = 0; j0 < T_SEQ; j0 += 32) {
    const u16* Kp = Kb + ((size_t)h * T_SEQ + j0) * DQK;
    f32x8 s0 = {}, s1 = {};
#pragma unroll
    for (int kk = 0; kk < 4; ++kk) {
      s0 = wmma_bf16(qf[kk], load_frag_b(Kp + kk * 32, DQK, lane), s0);
      s1 = wmma_bf16(qf[kk], load_frag_b(Kp + (size_t)16 * DQK + kk * 32, DQK, lane), s1);
    }
    float tmax[8];
#pragma unroll
    for (int i = 0; i < 8; ++i) {
      s0[i] *= scale; s1[i] *= scale;
      tmax[i] = fmaxf(s0[i], s1[i]);
    }
#pragma unroll
    for (int off = 1; off < 16; off <<= 1)
#pragma unroll
      for (int i = 0; i < 8; ++i)
        tmax[i] = fmaxf(tmax[i], __shfl_xor(tmax[i], off, 32));
    float alpha[8], rs[8];
#pragma unroll
    for (int i = 0; i < 8; ++i) {
      float mn = fmaxf(m_i[i], tmax[i]);
      alpha[i] = __expf(m_i[i] - mn);
      m_i[i] = mn;
      float p0 = __expf(s0[i] - mn);
      float p1 = __expf(s1[i] - mn);
      s0[i] = p0; s1[i] = p1;
      rs[i] = p0 + p1;
    }
#pragma unroll
    for (int off = 1; off < 16; off <<= 1)
#pragma unroll
      for (int i = 0; i < 8; ++i) rs[i] += __shfl_xor(rs[i], off, 32);
#pragma unroll
    for (int i = 0; i < 8; ++i) {
      l_i[i] = l_i[i] * alpha[i] + rs[i];
      o0[i] *= alpha[i]; o1[i] *= alpha[i]; o2[i] *= alpha[i]; o3[i] *= alpha[i];
    }
    // stage P (16x32) through this wave's private LDS tile to convert the
    // accumulator layout into an A-operand fragment (wave-internal, DS in-order)
#pragma unroll
    for (int i = 0; i < 8; ++i) {
      int r = i + 8 * g;
      ps[r * 32 + cc]      = f32_to_bf16(s0[i]);
      ps[r * 32 + 16 + cc] = f32_to_bf16(s1[i]);
    }
    asm volatile("s_wait_dscnt 0" ::: "memory");
    bf16x16 pa = load_frag_a(ps, 32, lane);
    const u16* Vp = Vt + ((size_t)h * DH) * T_SEQ + j0;
    o0 = wmma_bf16(pa, load_frag_b(Vp + (size_t)0  * T_SEQ, T_SEQ, lane), o0);
    o1 = wmma_bf16(pa, load_frag_b(Vp + (size_t)16 * T_SEQ, T_SEQ, lane), o1);
    o2 = wmma_bf16(pa, load_frag_b(Vp + (size_t)32 * T_SEQ, T_SEQ, lane), o2);
    o3 = wmma_bf16(pa, load_frag_b(Vp + (size_t)48 * T_SEQ, T_SEQ, lane), o3);
    asm volatile("s_wait_dscnt 0" ::: "memory");
  }
#pragma unroll
  for (int i = 0; i < 8; ++i) {
    float inv_l = 1.0f / l_i[i];
    int r = q0 + i + 8 * g;
    size_t base = (size_t)r * (NH * DH) + (size_t)h * DH + cc;
    Ob[base + 0]  = f32_to_bf16(o0[i] * inv_l);
    Ob[base + 16] = f32_to_bf16(o1[i] * inv_l);
    Ob[base + 32] = f32_to_bf16(o2[i] * inv_l);
    Ob[base + 48] = f32_to_bf16(o3[i] * inv_l);
  }
}

// ---------------------------------------------------------------------------
extern "C" void kernel_launch(void* const* d_in, const int* in_sizes, int n_in,
                              void* d_out, int out_size, void* d_ws, size_t ws_size,
                              hipStream_t stream) {
  (void)in_sizes; (void)n_in; (void)out_size; (void)ws_size;
  const float* x     = (const float*)d_in[0];
  const float* W_dq  = (const float*)d_in[1];
  const float* W_uq  = (const float*)d_in[2];
  const float* W_dkv = (const float*)d_in[3];
  const float* W_uk  = (const float*)d_in[4];
  const float* W_uv  = (const float*)d_in[5];
  const float* W_qr  = (const float*)d_in[6];
  const float* W_kr  = (const float*)d_in[7];
  const float* W_o   = (const float*)d_in[8];
  const float* qnw   = (const float*)d_in[9];
  const float* kvnw  = (const float*)d_in[10];

  float* y_out   = (float*)d_out;                          // 2048 x 2048
  float* ckv_out = y_out + (size_t)T_SEQ * DM;             // 2048 x 512
  float* krc_out = ckv_out + (size_t)T_SEQ * DC;           // 64 x 32

  char* ws = (char*)d_ws;
  size_t off = 0;
  auto alloc = [&](size_t bytes) -> char* {
    char* p = ws + off;
    off += (bytes + 255) & ~(size_t)255;
    return p;
  };
  u16* xb    = (u16*)alloc((size_t)T_SEQ * DM  * 2);
  u16* wdq   = (u16*)alloc((size_t)DCQ  * DM   * 2);
  u16* wuq   = (u16*)alloc((size_t)DM   * DCQ  * 2);
  u16* wdkv  = (u16*)alloc((size_t)DC   * DM   * 2);
  u16* wuk   = (u16*)alloc((size_t)DM   * DC   * 2);
  u16* wuv   = (u16*)alloc((size_t)DM   * DC   * 2);
  u16* wqr   = (u16*)alloc((size_t)64   * DCQ  * 2);
  u16* wkr   = (u16*)alloc((size_t)64   * DM   * 2);
  u16* wo    = (u16*)alloc((size_t)DM   * DM   * 2);
  float* cq_raw  = (float*)alloc((size_t)T_SEQ * DCQ * 4);
  float* ckv_raw = (float*)alloc((size_t)T_SEQ * DC  * 4);
  u16* cqb   = (u16*)alloc((size_t)T_SEQ * DCQ * 2);
  u16* ckvb  = (u16*)alloc((size_t)T_SEQ * DC  * 2);
  u16* Qb    = (u16*)alloc((size_t)NH * T_SEQ * DQK * 2);
  u16* Kb    = (u16*)alloc((size_t)NH * T_SEQ * DQK * 2);
  u16* Vt    = (u16*)alloc((size_t)NH * DH * T_SEQ * 2);
  // aliases (lifetimes don't overlap): Ob reuses cq_raw, qr/kr reuse ckv_raw
  u16*   Ob   = (u16*)cq_raw;                 // 2048*2048*2 <= 2048*1536*4
  float* qr_f = ckv_raw;                      // 2048*64*4
  float* kr_f = ckv_raw + (size_t)T_SEQ * 64; // 2048*64*4 (both fit in ckv_raw)

  auto cvt = [&](const float* src, u16* dst, size_t n) {
    cvt_bf16_kernel<<<dim3((unsigned)((n + 255) / 256)), dim3(256), 0, stream>>>(
        src, dst, (int)n);
  };
  // wide (128x128 block tile) GEMM for large N
  auto gemm4 = [&](const u16* A, const u16* W, float* Cf, u16* Cb,
                   int M, int N, int K, int mode) {
    gemm_bf16_kernel<4><<<dim3(N / 128, M / 128), dim3(256), 0, stream>>>(
        A, W, Cf, Cb, M, N, K, mode);
  };
  // narrow (128x64 block tile) GEMM for N=64 rope projections
  auto gemm2 = [&](const u16* A, const u16* W, float* Cf, u16* Cb,
                   int M, int N, int K, int mode) {
    gemm_bf16_kernel<2><<<dim3(N / 64, M / 128), dim3(256), 0, stream>>>(
        A, W, Cf, Cb, M, N, K, mode);
  };

  // 1) fp32 -> bf16
  cvt(x,     xb,   (size_t)T_SEQ * DM);
  cvt(W_dq,  wdq,  (size_t)DCQ * DM);
  cvt(W_uq,  wuq,  (size_t)DM * DCQ);
  cvt(W_dkv, wdkv, (size_t)DC * DM);
  cvt(W_uk,  wuk,  (size_t)DM * DC);
  cvt(W_uv,  wuv,  (size_t)DM * DC);
  cvt(W_qr,  wqr,  (size_t)64 * DCQ);
  cvt(W_kr,  wkr,  (size_t)64 * DM);
  cvt(W_o,   wo,   (size_t)DM * DM);

  // 2) down projections + RMSNorm
  gemm4(xb, wdq,  cq_raw,  nullptr, T_SEQ, DCQ, DM, 0);
  gemm4(xb, wdkv, ckv_raw, nullptr, T_SEQ, DC,  DM, 0);
  rmsnorm_kernel<<<dim3(T_SEQ), dim3(256), 0, stream>>>(cq_raw,  qnw,  nullptr, cqb,  DCQ);
  rmsnorm_kernel<<<dim3(T_SEQ), dim3(256), 0, stream>>>(ckv_raw, kvnw, ckv_out, ckvb, DC);

  // 3) up projections into attention layouts
  gemm4(cqb,  wuq, nullptr, Qb, T_SEQ, DM, DCQ, 1);  // q_content -> Q'[h][t][0:64]
  gemm4(ckvb, wuk, nullptr, Kb, T_SEQ, DM, DC,  1);  // k_content -> K'[h][t][0:64]
  gemm4(ckvb, wuv, nullptr, Vt, T_SEQ, DM, DC,  2);  // v         -> V^T[h][d][t]
  gemm2(cqb, wqr, qr_f, nullptr, T_SEQ, 64, DCQ, 0); // q_rope pre-rotation
  gemm2(xb,  wkr, kr_f, nullptr, T_SEQ, 64, DM,  0); // k_rope pre-rotation

  // 4) RoPE into rope half of Q'/K' (broadcast over heads); k_rope_cache out
  rope_kernel<<<dim3(T_SEQ), dim3(32), 0, stream>>>(qr_f, Qb, nullptr);
  rope_kernel<<<dim3(T_SEQ), dim3(32), 0, stream>>>(kr_f, Kb, krc_out);

  // 5) flash attention (d=128, dv=64, full softmax, no mask)
  flash_attn_kernel<<<dim3(T_SEQ / 128, NH), dim3(256), 0, stream>>>(Qb, Kb, Vt, Ob);

  // 6) output projection
  gemm4(Ob, wo, y_out, nullptr, T_SEQ, DM, DM, 0);
}
